// BatchHardTripletLoss_69982197121267
// MI455X (gfx1250) — matrix operand order repaired and checked by reading, not compiled, not run on hardware
//
#include <hip/hip_runtime.h>

// BatchHardTripletLoss for a,p,n: [4096,128] f32 -> scalar f32.
// Gram matrix via f16 hi/lo split (3x V_WMMA_F32_16X16X32_F16 per K-chunk),
// operands pre-converted to f16 once, tiles streamed to LDS with
// GLOBAL_LOAD_ASYNC_TO_LDS_B128 (ASYNCcnt), upper-triangular tiles only.

#define TB     4096          // triples (batch)
#define N3     12288         // 3*TB rows
#define D      128
#define TILE   128           // tile edge
#define KC     32            // K chunk per LDS pass
#define LROW   56            // LDS row stride in halves (KC + 24 pad; 112B = 16B multiple)
#define NTI    96            // N3 / TILE
#define NTRI   4656          // NTI*(NTI+1)/2
#define MARGIN 0.4f
#define ALPHA  0.01f

typedef __attribute__((ext_vector_type(16))) _Float16 v16h;
typedef __attribute__((ext_vector_type(8)))  _Float16 v8h;
typedef __attribute__((ext_vector_type(4)))  _Float16 v4h;
typedef __attribute__((ext_vector_type(8)))  float    v8f;

#define CAT16(lo, hi) __builtin_shufflevector(lo, hi, 0,1,2,3,4,5,6,7,8,9,10,11,12,13,14,15)

__device__ __forceinline__ const float* row_seg(const float* a, const float* p,
                                                const float* n, int base) {
  // base is a multiple of 128, so a 128-row block never straddles segments.
  return base < TB      ? a + (size_t)base * D
       : base < 2 * TB  ? p + (size_t)(base - TB) * D
                        : n + (size_t)(base - 2 * TB) * D;
}

// ---------------------------------------------------------------- init
__global__ void k_init(int* __restrict__ negmin, float* __restrict__ regsum) {
  int i = blockIdx.x * blockDim.x + threadIdx.x;
  if (i < N3) negmin[i] = 0x7F800000;  // +inf bits
  if (i == 0) *regsum = 0.0f;
}

// ---------------------------- one-time fp32 -> f16 hi/lo split of emb
__global__ __launch_bounds__(256) void k_convert(const float* __restrict__ a,
                                                 const float* __restrict__ p,
                                                 const float* __restrict__ n,
                                                 _Float16* __restrict__ ehi,
                                                 _Float16* __restrict__ elo) {
  int idx = blockIdx.x * 256 + threadIdx.x;   // float4 index; grid covers N3*D/4
  int row = idx >> 5;                         // 32 float4 per row
  const float* rp = row_seg(a, p, n, row & ~127) + (size_t)(row & 127) * D;
  float4 v = *((const float4*)rp + (idx & 31));
  float x[4] = {v.x, v.y, v.z, v.w};
  v4h h, l;
#pragma unroll
  for (int q = 0; q < 4; ++q) {
    _Float16 hh = (_Float16)x[q];             // RNE
    h[q] = hh;
    l[q] = (_Float16)(x[q] - (float)hh);      // residual, ~2^-21 total
  }
  *(v4h*)(ehi + (size_t)idx * 4) = h;
  *(v4h*)(elo + (size_t)idx * 4) = l;
}

// ------------------------------------------------- per-row ||x||^2 + reg
__global__ __launch_bounds__(256) void k_rowstats(const float* __restrict__ a,
                                                  const float* __restrict__ p,
                                                  const float* __restrict__ n,
                                                  float* __restrict__ sq,
                                                  float* __restrict__ regsum) {
  __shared__ float red[256];
  const int tid = threadIdx.x;
  const int i = blockIdx.x * 256 + tid;  // < N3 with grid 48
  const float* rp = row_seg(a, p, n, i & ~127) + (size_t)(i & 127) * D;
  const float4* v4 = (const float4*)rp;
  float s = 0.0f, rg = 0.0f;
#pragma unroll 8
  for (int j = 0; j < D / 4; ++j) {
    float4 v = v4[j];
    s += v.x * v.x + v.y * v.y + v.z * v.z + v.w * v.w;
    float t0 = fabsf(v.x) - 1.0f, t1 = fabsf(v.y) - 1.0f;
    float t2 = fabsf(v.z) - 1.0f, t3 = fabsf(v.w) - 1.0f;
    rg += t0 * t0 + t1 * t1 + t2 * t2 + t3 * t3;
  }
  sq[i] = s;
  red[tid] = rg;
  __syncthreads();
  for (int off = 128; off > 0; off >>= 1) {
    if (tid < off) red[tid] += red[tid + off];
    __syncthreads();
  }
  if (tid == 0) atomicAdd(regsum, red[0]);
}

// -------------------------------------------- exact intra-triple pos_max
__global__ __launch_bounds__(256) void k_posmax(const float* __restrict__ a,
                                                const float* __restrict__ p,
                                                const float* __restrict__ n,
                                                float* __restrict__ posmax) {
  int t = blockIdx.x * blockDim.x + threadIdx.x;
  if (t >= TB) return;
  const float4* av = (const float4*)(a + (size_t)t * D);
  const float4* pv = (const float4*)(p + (size_t)t * D);
  const float4* nv = (const float4*)(n + (size_t)t * D);
  float sap = 0.0f, san = 0.0f, spn = 0.0f;
#pragma unroll 8
  for (int j = 0; j < D / 4; ++j) {
    float4 va = av[j], vp = pv[j], vn = nv[j];
    float d;
    d = va.x - vp.x; sap += d * d;  d = va.y - vp.y; sap += d * d;
    d = va.z - vp.z; sap += d * d;  d = va.w - vp.w; sap += d * d;
    d = va.x - vn.x; san += d * d;  d = va.y - vn.y; san += d * d;
    d = va.z - vn.z; san += d * d;  d = va.w - vn.w; san += d * d;
    d = vp.x - vn.x; spn += d * d;  d = vp.y - vn.y; spn += d * d;
    d = vp.z - vn.z; spn += d * d;  d = vp.w - vn.w; spn += d * d;
  }
  float dap = sqrtf(sap), dan = sqrtf(san), dpn = sqrtf(spn);
  posmax[t]          = fmaxf(dap, dan);   // row a_t: positives {p_t, n_t}
  posmax[t + TB]     = fmaxf(dap, dpn);   // row p_t
  posmax[t + 2 * TB] = fmaxf(dan, dpn);   // row n_t
}

// -------- Gram tiles: async f16 tile fill + WMMA hi/lo + masked mins
__global__ __launch_bounds__(256) void k_gram(const _Float16* __restrict__ ehi,
                                              const _Float16* __restrict__ elo,
                                              const float* __restrict__ sq,
                                              int* __restrict__ negmin) {
  __shared__ __align__(16) _Float16 sAhi[TILE * LROW];
  __shared__ __align__(16) _Float16 sAlo[TILE * LROW];
  __shared__ __align__(16) _Float16 sBhi[TILE * LROW];
  __shared__ __align__(16) _Float16 sBlo[TILE * LROW];
  __shared__ float sSqR[TILE], sSqC[TILE];
  __shared__ int sRowMin[TILE], sColMin[TILE];

  const int tid = threadIdx.x;
  const int lane = tid & 31;
  const int wave = tid >> 5;       // 0..7: rows [wave*16, wave*16+16)
  const int lhalf = lane >> 4;     // 0 | 1
  const int lm = lane & 15;

  // decode upper-triangular tile (tr, tc), tc >= tr
  int t = blockIdx.x, tr = 0, rowlen = NTI;
  while (t >= rowlen) { t -= rowlen; --rowlen; ++tr; }
  const int tc = tr + t;
  const int rb = tr * TILE, cb = tc * TILE;

  // SGPR base addresses for async tile fill (GVS: saddr + 32-bit voffset)
  const unsigned long long ahiB = (unsigned long long)(uintptr_t)(ehi + (size_t)rb * D);
  const unsigned long long aloB = (unsigned long long)(uintptr_t)(elo + (size_t)rb * D);
  const unsigned long long bhiB = (unsigned long long)(uintptr_t)(ehi + (size_t)cb * D);
  const unsigned long long bloB = (unsigned long long)(uintptr_t)(elo + (size_t)cb * D);
  const unsigned lAhi = (unsigned)(uintptr_t)&sAhi[0];
  const unsigned lAlo = (unsigned)(uintptr_t)&sAlo[0];
  const unsigned lBhi = (unsigned)(uintptr_t)&sBhi[0];
  const unsigned lBlo = (unsigned)(uintptr_t)&sBlo[0];

  if (tid < TILE) {
    sSqR[tid] = sq[rb + tid];
    sSqC[tid] = sq[cb + tid];
    sRowMin[tid] = 0x7F800000;
    sColMin[tid] = 0x7F800000;
  }

  v8f acc[8];
  const v8f vzero = {0.f, 0.f, 0.f, 0.f, 0.f, 0.f, 0.f, 0.f};
#pragma unroll
  for (int c = 0; c < 8; ++c) acc[c] = vzero;

  for (int kp = 0; kp < D / KC; ++kp) {
    __syncthreads();  // previous pass's LDS reads are complete past this point
    // ---- async fill: 128 x KC halves per array, 16B chunks, ASYNCcnt-tracked
#pragma unroll
    for (int v = 0; v < 2; ++v) {
      int cidx = v * 256 + tid;            // 0..511 (4 chunks per row)
      int row = cidx >> 2;
      int kq = (cidx & 3) << 3;            // half offset 0,8,16,24
      unsigned goff = (unsigned)((row * D + kp * KC + kq) * 2);   // bytes
      unsigned loff = (unsigned)((row * LROW + kq) * 2);          // bytes
      asm volatile("global_load_async_to_lds_b128 %0, %1, %2"
                   :: "v"(lAhi + loff), "v"(goff), "s"(ahiB) : "memory");
      asm volatile("global_load_async_to_lds_b128 %0, %1, %2"
                   :: "v"(lAlo + loff), "v"(goff), "s"(aloB) : "memory");
      asm volatile("global_load_async_to_lds_b128 %0, %1, %2"
                   :: "v"(lBhi + loff), "v"(goff), "s"(bhiB) : "memory");
      asm volatile("global_load_async_to_lds_b128 %0, %1, %2"
                   :: "v"(lBlo + loff), "v"(goff), "s"(bloB) : "memory");
    }
    asm volatile("s_wait_asynccnt 0x0" ::: "memory");
    __syncthreads();

    // ---- A fragments for this wave's 16 rows (ISA 16-bit A 16x32 layout)
    const _Float16* pAh = &sAhi[(wave * 16 + lm) * LROW];
    const _Float16* pAl = &sAlo[(wave * 16 + lm) * LROW];
    v16h ahi = CAT16(*(const v8h*)(pAh + lhalf * 8), *(const v8h*)(pAh + 16 + lhalf * 8));
    v16h alo = CAT16(*(const v8h*)(pAl + lhalf * 8), *(const v8h*)(pAl + 16 + lhalf * 8));

#pragma unroll
    for (int c = 0; c < 8; ++c) {
      const _Float16* pBh = &sBhi[(c * 16 + lm) * LROW];
      const _Float16* pBl = &sBlo[(c * 16 + lm) * LROW];
      // B layout: same K order as A (8 K per half-wave, then +16)
      v16h bhi = CAT16(*(const v8h*)(pBh + lhalf * 8), *(const v8h*)(pBh + 16 + lhalf * 8));
      v16h blo = CAT16(*(const v8h*)(pBl + lhalf * 8), *(const v8h*)(pBl + 16 + lhalf * 8));
      // G = Ahi*Bhi + Ahi*Blo + Alo*Bhi  (lo*lo term negligible, ~2^-22)
      acc[c] = __builtin_amdgcn_wmma_f32_16x16x32_f16(false, ahi, false, bhi,
                                                      (short)0, acc[c], false, false);
      acc[c] = __builtin_amdgcn_wmma_f32_16x16x32_f16(false, ahi, false, blo,
                                                      (short)0, acc[c], false, false);
      acc[c] = __builtin_amdgcn_wmma_f32_16x16x32_f16(false, alo, false, bhi,
                                                      (short)0, acc[c], false, false);
    }
  }

  // ---- distances + masked min reductions
  const int rmod = rb & (TB - 1);  // rb % 4096 (no wrap within a tile)
  const int cmod = cb & (TB - 1);
  float rowmin[8];
#pragma unroll
  for (int r = 0; r < 8; ++r) rowmin[r] = INFINITY;

#pragma unroll
  for (int c = 0; c < 8; ++c) {
    const int col_local = c * 16 + lm;
    float colmin = INFINITY;
#pragma unroll
    for (int r = 0; r < 8; ++r) {
      const int row_local = wave * 16 + r + lhalf * 8;  // C/D layout
      float d2 = sSqR[row_local] + sSqC[col_local] - 2.0f * acc[c][r];
      float d = sqrtf(fmaxf(d2, 0.0f));
      // same-label mask: (global_row % 4096) == (global_col % 4096)
      if ((rmod + row_local) == (cmod + col_local)) d = INFINITY;
      colmin = fminf(colmin, d);
      rowmin[r] = fminf(rowmin[r], d);
    }
    atomicMin(&sColMin[col_local], __float_as_int(colmin));
  }
#pragma unroll
  for (int r = 0; r < 8; ++r) {
    float v = rowmin[r];
    v = fminf(v, __shfl_xor(v, 1, 32));
    v = fminf(v, __shfl_xor(v, 2, 32));
    v = fminf(v, __shfl_xor(v, 4, 32));
    v = fminf(v, __shfl_xor(v, 8, 32));  // min over 16-lane half
    if (lm == 0)
      atomicMin(&sRowMin[wave * 16 + r + lhalf * 8], __float_as_int(v));
  }
  __syncthreads();

  if (tid < TILE) {
    atomicMin(&negmin[rb + tid], sRowMin[tid]);
    if (tr != tc) atomicMin(&negmin[cb + tid], sColMin[tid]);  // symmetric half
  }
}

// ---------------------------------------------------------------- final
__global__ __launch_bounds__(256) void k_final(const float* __restrict__ posmax,
                                               const int* __restrict__ negmin,
                                               const float* __restrict__ regsum,
                                               float* __restrict__ out) {
  __shared__ float red[256];
  const int tid = threadIdx.x;
  float s = 0.0f;
  for (int i = tid; i < N3; i += 256) {
    float v = posmax[i] - __int_as_float(negmin[i]) + MARGIN;
    s += fmaxf(v, 0.0f);
  }
  red[tid] = s;
  __syncthreads();
  for (int off = 128; off > 0; off >>= 1) {
    if (tid < off) red[tid] += red[tid + off];
    __syncthreads();
  }
  if (tid == 0)
    out[0] = red[0] / (float)N3 + ALPHA * (regsum[0] / ((float)N3 * (float)D));
}

// ------------------------------------------------------------- launcher
extern "C" void kernel_launch(void* const* d_in, const int* in_sizes, int n_in,
                              void* d_out, int out_size, void* d_ws, size_t ws_size,
                              hipStream_t stream) {
  const float* a = (const float*)d_in[0];
  const float* p = (const float*)d_in[1];
  const float* n = (const float*)d_in[2];

  // ws layout: ehi[N3*D] f16 | elo[N3*D] f16 | negmin[N3] | sq[N3] | posmax[N3] | regsum
  _Float16* ehi = (_Float16*)d_ws;
  _Float16* elo = ehi + (size_t)N3 * D;
  int* negmin   = (int*)(elo + (size_t)N3 * D);
  float* sq     = (float*)(negmin + N3);
  float* posmax = (float*)(negmin + 2 * N3);
  float* regsum = (float*)(negmin + 3 * N3);

  k_init    <<<N3 / 256,       256, 0, stream>>>(negmin, regsum);
  k_convert <<<N3 * D / 1024,  256, 0, stream>>>(a, p, n, ehi, elo);
  k_rowstats<<<N3 / 256,       256, 0, stream>>>(a, p, n, sq, regsum);
  k_posmax  <<<TB / 256,       256, 0, stream>>>(a, p, n, posmax);
  k_gram    <<<NTRI,           256, 0, stream>>>(ehi, elo, sq, negmin);
  k_final   <<<1,              256, 0, stream>>>(posmax, negmin, regsum, (float*)d_out);
}
